// GCNModel3_45045617001060
// MI455X (gfx1250) — compile-verified, matrix operand order checked by reading, not run on hardware
//
#include <hip/hip_runtime.h>
#include <hip/hip_bf16.h>

// ---------------------------------------------------------------------------
// 2-layer GCN for MI455X (gfx1250, wave32).
//   - Dense node transforms via v_wmma_f32_16x16x32_f16 (f16 in, f32 acc).
//   - Weights pre-packed into WMMA B-fragment layout -> coalesced b128 loads
//     (replaces 16x global_load_u16 per k-step with 2x global_load_b128).
//   - Edge aggregation via global f32 atomics into an L2-resident (25.6MB)
//     accumulator (fits easily in the 192MB L2).
//   - All intermediates live in d_ws; re-initialized each call (graph-safe).
// ---------------------------------------------------------------------------

typedef _Float16 half_t;
typedef _Float16 v16h __attribute__((ext_vector_type(16)));
typedef _Float16 v8h  __attribute__((ext_vector_type(8)));
typedef float    v8f  __attribute__((ext_vector_type(8)));

#define IN_FEATS 128
#define HID      64
#define NLAB     8
#define NTN      4          // HID/16 column tiles
#define FRAG     512        // halves per packed B fragment (32 lanes x 16)

// ---------------- utility kernels ----------------

__global__ void zero_f32(float* __restrict__ p, int n) {
    int i = blockIdx.x * blockDim.x + threadIdx.x;
    if (i < n) p[i] = 0.0f;
}

__global__ void degree_kernel(const int* __restrict__ src, const int* __restrict__ dst,
                              float* __restrict__ dout, float* __restrict__ din, int E) {
    int e = blockIdx.x * blockDim.x + threadIdx.x;
    if (e < E) {
        atomicAdd(&dout[src[e]], 1.0f);
        atomicAdd(&din[dst[e]], 1.0f);
    }
}

__global__ void norm_kernel(const float* __restrict__ dout, const float* __restrict__ din,
                            float* __restrict__ ns, float* __restrict__ nd, int N) {
    int i = blockIdx.x * blockDim.x + threadIdx.x;
    if (i < N) {
        ns[i] = rsqrtf(fmaxf(dout[i], 1.0f));
        nd[i] = rsqrtf(fmaxf(din[i], 1.0f));
    }
}

// Pack W[K,64] (f32, row-major) into WMMA B-fragment layout (f16):
//   Bp[(kc*NTN + tn)*FRAG + lane*16 + i] = W[k(lane,i,kc)][tn*16 + (lane&15)]
// where, per CDNA5 ISA 7.12.2 (16-bit B operand, wave32):
//   base = kc*32 + (lane>>4)*8 ;  k = base + i (i<8),  base + 8 + i (i>=8)
__global__ void pack_b(const float* __restrict__ W, half_t* __restrict__ Bp, int kchunks) {
    int idx = blockIdx.x * blockDim.x + threadIdx.x;
    int total = kchunks * NTN * FRAG;
    if (idx >= total) return;
    int kc   = idx / (NTN * FRAG);
    int rest = idx % (NTN * FRAG);
    int tn   = rest / FRAG;
    int r    = rest % FRAG;
    int l    = r >> 4;
    int i    = r & 15;
    int k    = kc * 32 + (l >> 4) * 8 + ((i < 8) ? i : (8 + i));
    int n    = tn * 16 + (l & 15);
    Bp[idx] = (half_t)W[k * HID + n];
}

// ---------------- WMMA GEMMs ----------------
// One wave per 16x16 output tile. EXEC is all-1s (guard is wave-uniform).
// A fragment per ISA 7.12.2: lanes 0-15: K=base+0..7, base+16..23;
//                            lanes 16-31: K=base+8..15, base+24..31.
// C/D: reg g, lane l -> row = g + 8*(l>>4), col = l&15.

// H[N,64] = (feat[N,128] * nsrc[row]) @ W1  (B pre-packed)
__global__ void gemm1_wmma(const float* __restrict__ feat, const float* __restrict__ ns,
                           const half_t* __restrict__ Bp, float* __restrict__ H, int ntiles) {
    int wave = (blockIdx.x * blockDim.x + threadIdx.x) >> 5;
    if (wave >= ntiles) return;              // wave-uniform
    int lane = threadIdx.x & 31;
    int tm    = (wave >> 2) << 4;
    int tnIdx = wave & 3;
    int l15 = lane & 15;
    int hs  = lane >> 4;
    int mrow = tm + l15;
    float sc = ns[mrow];
    const float* arow = feat + (long)mrow * IN_FEATS;
    const v16h*  bfrag = (const v16h*)Bp;    // 32 v16h per (kc,tn) fragment
    v8f c = {};
    #pragma unroll
    for (int kc = 0; kc < IN_FEATS / 32; ++kc) {
        int base = kc * 32 + hs * 8;
        float4 f0 = *(const float4*)(arow + base);
        float4 f1 = *(const float4*)(arow + base + 4);
        float4 f2 = *(const float4*)(arow + base + 16);
        float4 f3 = *(const float4*)(arow + base + 20);
        v16h a;
        a[0]  = (half_t)(f0.x * sc); a[1]  = (half_t)(f0.y * sc);
        a[2]  = (half_t)(f0.z * sc); a[3]  = (half_t)(f0.w * sc);
        a[4]  = (half_t)(f1.x * sc); a[5]  = (half_t)(f1.y * sc);
        a[6]  = (half_t)(f1.z * sc); a[7]  = (half_t)(f1.w * sc);
        a[8]  = (half_t)(f2.x * sc); a[9]  = (half_t)(f2.y * sc);
        a[10] = (half_t)(f2.z * sc); a[11] = (half_t)(f2.w * sc);
        a[12] = (half_t)(f3.x * sc); a[13] = (half_t)(f3.y * sc);
        a[14] = (half_t)(f3.z * sc); a[15] = (half_t)(f3.w * sc);
        v16h b = bfrag[(kc * NTN + tnIdx) * 32 + lane];   // coalesced 32B/lane
        c = __builtin_amdgcn_wmma_f32_16x16x32_f16(false, a, false, b, (short)0, c, false, false);
    }
    int ncol = (tnIdx << 4) + l15;
    #pragma unroll
    for (int g = 0; g < 8; ++g) {
        int row = tm + g + hs * 8;
        H[(long)row * HID + ncol] = c[g];
    }
}

// H[N,64] = A2h[N,64] @ W2  (A already f16, pre-scaled by norm_src; B packed)
__global__ void gemm2_wmma(const half_t* __restrict__ A, const half_t* __restrict__ Bp,
                           float* __restrict__ H, int ntiles) {
    int wave = (blockIdx.x * blockDim.x + threadIdx.x) >> 5;
    if (wave >= ntiles) return;
    int lane = threadIdx.x & 31;
    int tm    = (wave >> 2) << 4;
    int tnIdx = wave & 3;
    int l15 = lane & 15;
    int hs  = lane >> 4;
    int mrow = tm + l15;
    const half_t* arow = A + (long)mrow * HID;
    const v16h*  bfrag = (const v16h*)Bp;
    v8f c = {};
    #pragma unroll
    for (int kc = 0; kc < HID / 32; ++kc) {
        int base = kc * 32 + hs * 8;
        v8h alo = *(const v8h*)(arow + base);        // 16B vector load
        v8h ahi = *(const v8h*)(arow + base + 16);   // 16B vector load
        v16h a = __builtin_shufflevector(alo, ahi,
                                         0, 1, 2, 3, 4, 5, 6, 7,
                                         8, 9, 10, 11, 12, 13, 14, 15);
        v16h b = bfrag[(kc * NTN + tnIdx) * 32 + lane];
        c = __builtin_amdgcn_wmma_f32_16x16x32_f16(false, a, false, b, (short)0, c, false, false);
    }
    int ncol = (tnIdx << 4) + l15;
    #pragma unroll
    for (int g = 0; g < 8; ++g) {
        int row = tm + g + hs * 8;
        H[(long)row * HID + ncol] = c[g];
    }
}

// ---------------- edge scatter-add (segment_sum) ----------------
// 16 threads per edge, float4 gather of h[src], 4 f32 atomics into agg[dst].
__global__ void scatter_add(const float* __restrict__ H, const int* __restrict__ src,
                            const int* __restrict__ dst, float* __restrict__ agg, int E) {
    int t = blockIdx.x * blockDim.x + threadIdx.x;
    int e = t >> 4;
    if (e >= E) return;
    int j = (t & 15) << 2;
    int s = src[e], d = dst[e];
    const float4 v = *reinterpret_cast<const float4*>(H + (long)s * HID + j);
    float* o = agg + (long)d * HID + j;
    atomicAdd(o + 0, v.x);
    atomicAdd(o + 1, v.y);
    atomicAdd(o + 2, v.z);
    atomicAdd(o + 3, v.w);
}

// ---------------- fused epilogue: relu(agg*nd + b1) * ns  ->  f16 A2 ----------------
__global__ void epilogue1(const float* __restrict__ agg, const float* __restrict__ nd,
                          const float* __restrict__ ns, const float* __restrict__ bias,
                          half_t* __restrict__ A2, int total) {
    int t = blockIdx.x * blockDim.x + threadIdx.x;
    if (t >= total) return;
    int n = t >> 6;
    int k = t & 63;
    float v = agg[t] * nd[n] + bias[k];
    v = fmaxf(v, 0.0f) * ns[n];
    A2[t] = (half_t)v;
}

// ---------------- head: x2 = relu(agg*nd + b2); softmax(x2 @ Wp + bp) ----------------
__global__ void head_kernel(const float* __restrict__ agg, const float* __restrict__ nd,
                            const float* __restrict__ b2, const float* __restrict__ Wp,
                            const float* __restrict__ bp, float* __restrict__ out, int N) {
    int wave = (blockIdx.x * blockDim.x + threadIdx.x) >> 5;
    int lane = threadIdx.x & 31;
    if (wave >= N) return;
    float ndv = nd[wave];
    float acc[NLAB];
    #pragma unroll
    for (int l = 0; l < NLAB; ++l) acc[l] = 0.0f;
    #pragma unroll
    for (int kk = lane; kk < HID; kk += 32) {
        float v = fmaxf(agg[(long)wave * HID + kk] * ndv + b2[kk], 0.0f);
        #pragma unroll
        for (int l = 0; l < NLAB; ++l) acc[l] += v * Wp[kk * NLAB + l];
    }
    #pragma unroll
    for (int l = 0; l < NLAB; ++l) {
        #pragma unroll
        for (int off = 16; off > 0; off >>= 1)
            acc[l] += __shfl_down(acc[l], off, 32);
    }
    if (lane == 0) {
        float m = -1e30f;
        #pragma unroll
        for (int l = 0; l < NLAB; ++l) { acc[l] += bp[l]; m = fmaxf(m, acc[l]); }
        float s = 0.0f;
        #pragma unroll
        for (int l = 0; l < NLAB; ++l) { acc[l] = __expf(acc[l] - m); s += acc[l]; }
        float inv = 1.0f / s;
        #pragma unroll
        for (int l = 0; l < NLAB; ++l) out[(long)wave * NLAB + l] = acc[l] * inv;
    }
}

// ---------------- launch ----------------

static inline int cdiv(long a, int b) { return (int)((a + b - 1) / b); }

extern "C" void kernel_launch(void* const* d_in, const int* in_sizes, int n_in,
                              void* d_out, int out_size, void* d_ws, size_t ws_size,
                              hipStream_t stream) {
    const float* feat = (const float*)d_in[0];
    const int*   esrc = (const int*)d_in[1];
    const int*   edst = (const int*)d_in[2];
    /* d_in[3] edge_types: unused by the reference */
    const float* W1 = (const float*)d_in[4];
    const float* b1 = (const float*)d_in[5];
    const float* W2 = (const float*)d_in[6];
    const float* b2 = (const float*)d_in[7];
    const float* Wp = (const float*)d_in[8];
    const float* bp = (const float*)d_in[9];
    float* out = (float*)d_out;

    const int N = in_sizes[0] / IN_FEATS;   // 100000 (divisible by 16)
    const int E = in_sizes[1];              // 1600000

    // Workspace layout (floats):
    float*  ws      = (float*)d_ws;
    float*  deg_out = ws;                    // [N]
    float*  deg_in  = ws + (long)N;          // [N]
    float*  nsrc    = ws + 2L * N;           // [N]
    float*  ndst    = ws + 3L * N;           // [N]
    float*  hbuf    = ws + 4L * N;           // [N*64]  (reused for both layers)
    float*  aggbuf  = ws + 68L * N;          // [N*64]  (reused, re-zeroed)
    half_t* A2h     = (half_t*)(ws + 132L * N); // N*64 halves (= 32N floats)
    half_t* B1p     = (half_t*)(ws + 164L * N); // 4*4*512 halves (packed W1)
    half_t* B2p     = B1p + (IN_FEATS / 32) * NTN * FRAG; // 2*4*512 halves
    // total: ~164N floats + 24KB  (~66 MB)

    const int B = 256;
    const int ntiles = (N / 16) * (HID / 16);   // 25000 waves per GEMM
    const int kc1 = IN_FEATS / 32, kc2 = HID / 32;

    // degrees + norms (shared by both layers) + weight fragment packing
    zero_f32<<<cdiv(2L * N, B), B, 0, stream>>>(deg_out, 2 * N);
    zero_f32<<<cdiv(64L * N, B), B, 0, stream>>>(aggbuf, 64 * N);
    degree_kernel<<<cdiv(E, B), B, 0, stream>>>(esrc, edst, deg_out, deg_in, E);
    norm_kernel<<<cdiv(N, B), B, 0, stream>>>(deg_out, deg_in, nsrc, ndst, N);
    pack_b<<<cdiv((long)kc1 * NTN * FRAG, B), B, 0, stream>>>(W1, B1p, kc1);
    pack_b<<<cdiv((long)kc2 * NTN * FRAG, B), B, 0, stream>>>(W2, B2p, kc2);

    // layer 1
    gemm1_wmma<<<cdiv((long)ntiles * 32, B), B, 0, stream>>>(feat, nsrc, B1p, hbuf, ntiles);
    scatter_add<<<cdiv((long)E * 16, B), B, 0, stream>>>(hbuf, esrc, edst, aggbuf, E);
    epilogue1<<<cdiv((long)N * HID, B), B, 0, stream>>>(aggbuf, ndst, nsrc, b1, A2h, N * HID);

    // layer 2
    zero_f32<<<cdiv(64L * N, B), B, 0, stream>>>(aggbuf, 64 * N);
    gemm2_wmma<<<cdiv((long)ntiles * 32, B), B, 0, stream>>>(A2h, B2p, hbuf, ntiles);
    scatter_add<<<cdiv((long)E * 16, B), B, 0, stream>>>(hbuf, esrc, edst, aggbuf, E);

    // head + softmax
    head_kernel<<<cdiv((long)N * 32, B), B, 0, stream>>>(aggbuf, ndst, b2, Wp, bp, out, N);
}
